// ConsineMultiHeadAttention_5428838662760
// MI455X (gfx1250) — matrix operand order repaired and checked by reading, not compiled, not run on hardware
//
#include <hip/hip_runtime.h>
#include <hip/hip_bf16.h>

// Problem constants (match reference)
constexpr int B_  = 4;
constexpr int S_  = 4096;
constexpr int D_  = 1024;
constexpr int HD_ = 120;
constexpr int NP_ = 128;        // HD padded to 128 (multiple of WMMA K=32)
constexpr int SROW_ = S_ + 4;   // LDS row stride (8*4100 % 64 == 32 -> halves hit disjoint banks)

typedef __attribute__((ext_vector_type(16))) __bf16 v16bf;
typedef __attribute__((ext_vector_type(8)))  float  v8f;

__device__ __forceinline__ __bf16 f2bf(float f) {
  unsigned u = __builtin_bit_cast(unsigned, f);
  u += 0x7FFFu + ((u >> 16) & 1u);                 // round-to-nearest-even
  unsigned short h = (unsigned short)(u >> 16);
  return __builtin_bit_cast(__bf16, h);
}
__device__ __forceinline__ unsigned short f2bfu(float f) {
  unsigned u = __builtin_bit_cast(unsigned, f);
  u += 0x7FFFu + ((u >> 16) & 1u);
  return (unsigned short)(u >> 16);
}
__device__ __forceinline__ void cvt4(v16bf& v, int base, float4 f) {
  v[base + 0] = f2bf(f.x); v[base + 1] = f2bf(f.y);
  v[base + 2] = f2bf(f.z); v[base + 3] = f2bf(f.w);
}

// ---------------------------------------------------------------------------
// Kernel 0: one-time Wq f32 -> bf16 conversion, padded to 128 rows (rows
// 120..127 zero). Removes ~800M redundant per-wave VALU conversions from the
// projection GEMM and kills the exec-mask branch in its K-loop.
// 128*1024 elements, 4 per thread -> 32768 threads.
// ---------------------------------------------------------------------------
__global__ __launch_bounds__(256) void wq_cvt_kernel(
    const float* __restrict__ Wq, unsigned short* __restrict__ Wqb)
{
  const int t    = blockIdx.x * 256 + threadIdx.x;   // 0..32767
  const int base = t * 4;                            // element index in [128*1024)
  const int n    = base >> 10;                       // row
  const int k    = base & 1023;                      // col
  ushort4 o;
  if (n < HD_) {
    float4 f = *(const float4*)(Wq + (size_t)n * D_ + k);
    o.x = f2bfu(f.x); o.y = f2bfu(f.y); o.z = f2bfu(f.z); o.w = f2bfu(f.w);
  } else {
    o.x = o.y = o.z = o.w = 0;
  }
  *(ushort4*)(Wqb + base) = o;
}

// ---------------------------------------------------------------------------
// Kernel A: ctx_n[b*S+s, 0..127] = normalize( (query @ Wq^T + bq) * wt )
// One wave per 16-row block; 8 accumulator tiles cover N=0..127 so the L2
// norm reduces inside the wave. WMMA bf16, fp32 accumulate.
// A-frag: lane<16 -> row m0+lane, K = kc+{0..7,16..23}; lane>=16 -> row
//         m0+lane-16, K = kc+{8..15,24..31}  (converted inline, once per row).
// B-frag: lane l -> 2x8 bf16 of Wqb row nt*16+(l&15) at
//         cols kc + (l>>4)*8 + {0..7,16..23}  (pre-converted, branch-free).
// ---------------------------------------------------------------------------
__global__ __launch_bounds__(256) void ctxn_gemm_kernel(
    const float* __restrict__ q, const unsigned short* __restrict__ Wqb,
    const float* __restrict__ bq, const float* __restrict__ wt,
    unsigned short* __restrict__ ctx)
{
  __shared__ unsigned short stg[8][16][NP_];   // 32 KB, one 16x128 tile per wave

  const int wave = threadIdx.x >> 5;
  const int lane = threadIdx.x & 31;
  const int l15  = lane & 15;
  const int hi   = lane >> 4;
  const int rb   = blockIdx.x * 8 + wave;      // 1024 row-blocks total
  const int m0   = rb * 16;

  v8f acc[8];
#pragma unroll
  for (int nt = 0; nt < 8; ++nt)
#pragma unroll
    for (int r = 0; r < 8; ++r) acc[nt][r] = 0.0f;

  const float* arow = q + (size_t)(m0 + l15) * D_;
  const unsigned short* brow = Wqb + (size_t)l15 * D_;   // + nt*16*D_ + kc + hi*8

  for (int kc = 0; kc < D_; kc += 32) {
    v16bf a;
    {
      const float* p1 = arow + kc + hi * 8;
      const float* p2 = arow + kc + 16 + hi * 8;
      cvt4(a, 0,  *(const float4*)(p1));
      cvt4(a, 4,  *(const float4*)(p1 + 4));
      cvt4(a, 8,  *(const float4*)(p2));
      cvt4(a, 12, *(const float4*)(p2 + 4));
    }
#pragma unroll
    for (int nt = 0; nt < 8; ++nt) {
      union { uint4 qd[2]; v16bf v; } u;
      const unsigned short* p = brow + (size_t)nt * 16 * D_ + kc + hi * 8;
      u.qd[0] = *(const uint4*)(p);
      u.qd[1] = *(const uint4*)(p + 16);
      acc[nt] = __builtin_amdgcn_wmma_f32_16x16x32_bf16(
          false, a, false, u.v, (short)0, acc[nt], false, false);
    }
  }

  // Epilogue: bias, channel scale, row L2-normalize, emit bf16.
  float wv[8], bv[8], ss[8];
#pragma unroll
  for (int r = 0; r < 8; ++r) ss[r] = 0.0f;
#pragma unroll
  for (int nt = 0; nt < 8; ++nt) {
    const int n = nt * 16 + l15;
    const bool v = (n < HD_);
    wv[nt] = v ? wt[n] : 0.0f;   // zeroes padded cols 120..127
    bv[nt] = v ? bq[n] : 0.0f;
  }
#pragma unroll
  for (int nt = 0; nt < 8; ++nt)
#pragma unroll
    for (int r = 0; r < 8; ++r) {
      float x = (acc[nt][r] + bv[nt]) * wv[nt];
      acc[nt][r] = x;
      ss[r] += x * x;
    }
  // reduce over N (16 lanes of each half-group); masks<16 stay within halves
#pragma unroll
  for (int r = 0; r < 8; ++r) {
    float s = ss[r];
    s += __shfl_xor(s, 1, 32);
    s += __shfl_xor(s, 2, 32);
    s += __shfl_xor(s, 4, 32);
    s += __shfl_xor(s, 8, 32);
    ss[r] = 1.0f / fmaxf(sqrtf(s), 1e-12f);   // F.normalize eps semantics
  }
  // D layout: VGPR r -> local row r + hi*8, col nt*16 + (lane&15)
#pragma unroll
  for (int nt = 0; nt < 8; ++nt)
#pragma unroll
    for (int r = 0; r < 8; ++r)
      stg[wave][hi * 8 + r][nt * 16 + l15] = f2bfu(acc[nt][r] * ss[r]);

  // Wave-private LDS staging -> coalesced b128 global stores (same-wave RAW,
  // compiler inserts s_wait_dscnt; no workgroup barrier needed).
  const int row  = lane >> 1;
  const int half = lane & 1;
#pragma unroll
  for (int i = 0; i < 8; ++i) {
    uint4 v = *(const uint4*)&stg[wave][row][half * 64 + i * 8];
    *(uint4*)(ctx + (size_t)(m0 + row) * NP_ + half * 64 + i * 8) = v;
  }
}

// ---------------------------------------------------------------------------
// Kernel B: fused Gram-matrix + row softmax.
// One 256-thread block (8 waves) per (batch, 16-row panel). Raw scores for
// 16 x 4096 buffered in 263 KB LDS (CDNA5: up to 320 KB per workgroup),
// so d_out is written exactly once (bandwidth roofline ~256 MB @ 23.3 TB/s).
// ---------------------------------------------------------------------------
__global__ __launch_bounds__(256) void attn_softmax_kernel(
    const unsigned short* __restrict__ ctx, float* __restrict__ out)
{
  extern __shared__ float smem[];
  float* sbuf  = smem;                    // 16 * SROW_
  float* red   = smem + 16 * SROW_;       // 256 partials
  float* rstat = red + 256;               // [0..15]=rowmax, [16..31]=1/sum

  const int b    = blockIdx.x >> 8;       // 256 panels per batch
  const int s0   = (blockIdx.x & 255) * 16;
  const int tid  = threadIdx.x;
  const int wave = tid >> 5;
  const int lane = tid & 31;
  const int l15  = lane & 15;
  const int hi   = lane >> 4;

  const unsigned short* base = ctx + (size_t)b * S_ * NP_;

  // Preload the 4 A-chunks (K = 128) for this 16-row panel.
  v16bf afrag[4];
  {
    const unsigned short* arow = base + (size_t)(s0 + l15) * NP_;
#pragma unroll
    for (int ch = 0; ch < 4; ++ch) {
      const int kc = ch * 32;
      union { uint4 qd[2]; v16bf v; } u;
      u.qd[0] = *(const uint4*)(arow + kc + hi * 8);
      u.qd[1] = *(const uint4*)(arow + kc + 16 + hi * 8);
      afrag[ch] = u.v;
    }
  }

  // Sweep all 256 key tiles; B-frag = 2x16 bytes per lane
  // (row t0+(lane&15), cols kc+hi*8..+7, kc+16+hi*8..+7) — Gram symmetry: no transpose.
  for (int i = wave; i < 256; i += 8) {
    const int t0 = i * 16;
    const unsigned short* brow = base + (size_t)(t0 + l15) * NP_;
    v8f c;
#pragma unroll
    for (int r = 0; r < 8; ++r) c[r] = 0.0f;
#pragma unroll
    for (int ch = 0; ch < 4; ++ch) {
      union { uint4 qd[2]; v16bf v; } u;
      const unsigned short* p = brow + ch * 32 + hi * 8;
      u.qd[0] = *(const uint4*)(p);
      u.qd[1] = *(const uint4*)(p + 16);
      c = __builtin_amdgcn_wmma_f32_16x16x32_bf16(
          false, afrag[ch], false, u.v, (short)0, c, false, false);
    }
    const int col = t0 + l15;
#pragma unroll
    for (int r = 0; r < 8; ++r)
      sbuf[(hi * 8 + r) * SROW_ + col] = c[r];
  }
  __syncthreads();

  // Softmax over 4096 per row; 16 threads per row, stride-16 interleave.
  const int row = tid >> 4;
  const int seg = tid & 15;
  {
    const float* rp = sbuf + row * SROW_ + seg;
    float mx = -3.402823466e38f;
#pragma unroll 4
    for (int j = 0; j < 256; ++j) mx = fmaxf(mx, rp[j * 16]);
    red[tid] = mx;
  }
  __syncthreads();
  if (tid < 16) {
    float m = red[tid * 16];
#pragma unroll
    for (int j = 1; j < 16; ++j) m = fmaxf(m, red[tid * 16 + j]);
    rstat[tid] = m;
  }
  __syncthreads();
  {
    const float rm = rstat[row];
    float* wp = sbuf + row * SROW_ + seg;
    float sum = 0.0f;
#pragma unroll 4
    for (int j = 0; j < 256; ++j) {
      float e = __expf(wp[j * 16] - rm);
      wp[j * 16] = e;
      sum += e;
    }
    red[tid] = sum;
  }
  __syncthreads();
  if (tid < 16) {
    float s = 0.0f;
#pragma unroll
    for (int j = 0; j < 16; ++j) s += red[tid * 16 + j];
    rstat[16 + tid] = 1.0f / s;   // max element contributes exp(0)=1 -> s>0
  }
  __syncthreads();

  // Single coalesced write of the final softmax panel.
#pragma unroll 1
  for (int r = 0; r < 16; ++r) {
    const float inv = rstat[16 + r];
    const float4* s4 = (const float4*)(sbuf + r * SROW_);  // 16 B aligned
    float4* o4 = (float4*)(out + ((size_t)b * S_ + s0 + r) * S_);
    for (int j = tid; j < S_ / 4; j += 256) {
      float4 v = s4[j];
      v.x *= inv; v.y *= inv; v.z *= inv; v.w *= inv;
      o4[j] = v;
    }
  }
}

// ---------------------------------------------------------------------------
// d_in order: query[B,S,D] f32, key (unused), Wq[HD,D] f32, bq[HD] f32,
//             weight_tensor[H,HD] f32.  d_out: [B,S,S] f32.
// d_ws: [0, 4MB)   bf16 ctx_n buffer [B*S, 128]
//       [4MB, +256KB) bf16 Wq buffer [128, 1024], rows 120..127 zero
// ---------------------------------------------------------------------------
extern "C" void kernel_launch(void* const* d_in, const int* in_sizes, int n_in,
                              void* d_out, int out_size, void* d_ws, size_t ws_size,
                              hipStream_t stream) {
  const float* q  = (const float*)d_in[0];
  const float* Wq = (const float*)d_in[2];
  const float* bq = (const float*)d_in[3];
  const float* wt = (const float*)d_in[4];
  float* out = (float*)d_out;

  unsigned short* ctx = (unsigned short*)d_ws;                    // bf16 bits
  unsigned short* Wqb = ctx + (size_t)(B_ * S_) * NP_;            // +4 MB

  // Stage 0: Wq -> bf16, zero-padded to 128 rows. 32768 threads.
  wq_cvt_kernel<<<dim3(128), dim3(256), 0, stream>>>(Wq, Wqb);

  // Stage A: 1024 row-blocks, 8 waves/block -> 128 blocks.
  ctxn_gemm_kernel<<<dim3(128), dim3(256), 0, stream>>>(q, Wqb, bq, wt, ctx);

  // Stage B: one block per (batch, 16-row panel) = 4*256 blocks.
  const size_t shbytes = (size_t)(16 * SROW_ + 256 + 32) * sizeof(float);
  attn_softmax_kernel<<<dim3(B_ * (S_ / 16)), dim3(256), shbytes, stream>>>(ctx, out);
}
